// Mamba_6828998001042
// MI455X (gfx1250) — matrix operand order repaired and checked by reading, not compiled, not run on hardware
//
#include <hip/hip_runtime.h>
#include <math.h>

// ---------------------------------------------------------------------------
// Mamba forward for MI455X (gfx1250, wave32).
//  - all GEMMs: v_wmma_f32_16x16x32_bf16 (bf16 operands, f32 accum)
//  - tile staging: global_load_async_to_lds_b128 (ASYNCcnt DMA), double-
//    buffered so tile k+1 DMA overlaps tile k WMMA work
//  - operands pre-converted fp32->bf16 (halves GEMM traffic; A-tile re-reads
//    hit the 192MB L2)
//  - selective scan: register-resident 16-state recurrence per channel,
//    LDS-staged B/C vectors, fused D-skip + SiLU gate, bf16 output
// ---------------------------------------------------------------------------

typedef __bf16 bf16;
typedef __attribute__((ext_vector_type(16))) __bf16 v16bf;
typedef __attribute__((ext_vector_type(8)))  __bf16 v8bf;
typedef __attribute__((ext_vector_type(8)))  float  v8f;

__device__ __forceinline__ unsigned short bf16bits(float f) {
    unsigned u = __builtin_bit_cast(unsigned, f);
    u += 0x7fffu + ((u >> 16) & 1u);            // round-to-nearest-even
    return (unsigned short)(u >> 16);
}
__device__ __forceinline__ bf16 f2bf(float f) {
    unsigned short h = bf16bits(f);
    return __builtin_bit_cast(bf16, h);
}
__device__ __forceinline__ float silu_f(float x) {
    return x / (1.f + __expf(-x));
}

// CDNA5 async memory->LDS DMA (16B per lane), tracked by ASYNCcnt.
__device__ __forceinline__ void async_load_b128(unsigned lds_off, const void* gptr) {
    asm volatile("global_load_async_to_lds_b128 %0, %1, off"
                 :: "v"(lds_off), "v"(gptr) : "memory");
}
__device__ __forceinline__ void wait_async0() {
    asm volatile("s_wait_asynccnt 0" ::: "memory");
}
// allow the 4 just-issued next-tile chunks to remain in flight
__device__ __forceinline__ void wait_async4() {
    asm volatile("s_wait_asynccnt 4" ::: "memory");
}

// ---------------------------------------------------------------------------
// fp32 -> bf16 bulk convert (float4 in, 2x packed bf16 out); n % 4 == 0
// ---------------------------------------------------------------------------
__global__ __launch_bounds__(256)
void cvt_f32_to_bf16(const float* __restrict__ s, unsigned* __restrict__ d, int n4)
{
    int i = blockIdx.x * blockDim.x + threadIdx.x;
    if (i >= n4) return;
    float4 f = ((const float4*)s)[i];
    unsigned lo = ((unsigned)bf16bits(f.y) << 16) | bf16bits(f.x);
    unsigned hi = ((unsigned)bf16bits(f.w) << 16) | bf16bits(f.z);
    uint2 v; v.x = lo; v.y = hi;
    ((uint2*)d)[i] = v;
}

// ---------------------------------------------------------------------------
// C[M,N] = A[M,K] @ B[N,K]^T  (bf16 operands, row-major, K contiguous)
// Requirements (all call sites satisfy them): M%128==0, K%32==0; only N ragged.
// mode 0: plain; mode 1: softplus(x + bias[n]).  C2 (optional): bf16 copy.
// ---------------------------------------------------------------------------
#define BM 128
#define BN 128
#define BK 32
#define LDP 40   // padded LDS row (bf16); 80B row keeps 16B chunk alignment

__global__ __launch_bounds__(256)
void gemm_bf16_wmma(const bf16* __restrict__ A, int lda,
                    const bf16* __restrict__ B, int ldb,
                    float* __restrict__ C, int ldc,
                    bf16* __restrict__ C2,
                    int M, int N, int K,
                    const float* __restrict__ bias, int mode)
{
    __shared__ bf16 sA[2][BM][LDP];   // double-buffered tiles
    __shared__ bf16 sB[2][BM][LDP];

    const int tid  = threadIdx.x;
    const int wave = tid >> 5;
    const int lane = tid & 31;
    const int lh   = lane >> 4;
    const int ll   = lane & 15;

    const int m0 = blockIdx.y * BM;
    const int n0 = blockIdx.x * BN;
    const int wm = (wave >> 2) * 64;   // 2 wave rows
    const int wn = (wave & 3) * 32;    // 4 wave cols

    v8f acc[4][2];
    #pragma unroll
    for (int i = 0; i < 4; ++i)
        #pragma unroll
        for (int j = 0; j < 2; ++j)
            #pragma unroll
            for (int r = 0; r < 8; ++r) acc[i][j][r] = 0.f;

    // staging map: per pass each thread moves one 8-elem (16B) chunk
    // (BM*BK/8 = 512 chunks, 2 passes of 256) -> 4 async issues per tile
    const int srow0 = tid >> 2;
    const int scol  = (tid & 3) * 8;
    const bf16* Abase = A + (size_t)(m0 + srow0) * lda + scol;
    const bf16* Bbase = B + (size_t)(n0 + srow0) * ldb + scol;

    auto issue_tile = [&](int kt, int buf) {
        size_t ko = (size_t)kt * BK;
        #pragma unroll
        for (int p = 0; p < 2; ++p) {
            int row = srow0 + p * 64;
            async_load_b128((unsigned)(size_t)&sA[buf][row][scol],
                            Abase + (size_t)p * 64 * lda + ko);
            async_load_b128((unsigned)(size_t)&sB[buf][row][scol],
                            Bbase + (size_t)p * 64 * ldb + ko);
        }
    };

    const int nk = K / BK;
    issue_tile(0, 0);

    for (int kt = 0; kt < nk; ++kt) {
        const int cur = kt & 1;
        if (kt + 1 < nk) {
            issue_tile(kt + 1, cur ^ 1);  // overlap next DMA with this tile's math
            wait_async4();                // current tile's 4 chunks complete
        } else {
            wait_async0();
        }
        __syncthreads();                  // tile resident for all waves

        v16bf afrag[4], bfrag[2];
        #pragma unroll
        for (int i = 0; i < 4; ++i) {
            int m  = wm + i * 16 + ll;
            int kb = 8 * lh;
            v8bf lo = *(const v8bf*)&sA[cur][m][kb];
            v8bf hi = *(const v8bf*)&sA[cur][m][kb + 16];
            afrag[i] = __builtin_shufflevector(lo, hi,
                0,1,2,3,4,5,6,7,8,9,10,11,12,13,14,15);
        }
        #pragma unroll
        for (int j = 0; j < 2; ++j) {
            int n  = wn + j * 16 + ll;
            int kc = 8 * lh;
            v8bf lo = *(const v8bf*)&sB[cur][n][kc];
            v8bf hi = *(const v8bf*)&sB[cur][n][kc + 16];
            bfrag[j] = __builtin_shufflevector(lo, hi,
                0,1,2,3,4,5,6,7,8,9,10,11,12,13,14,15);
        }

        #pragma unroll
        for (int i = 0; i < 4; ++i)
            #pragma unroll
            for (int j = 0; j < 2; ++j)
                acc[i][j] = __builtin_amdgcn_wmma_f32_16x16x32_bf16(
                    false, afrag[i], false, bfrag[j],
                    (short)0, acc[i][j], false, false);

        __syncthreads();   // all waves done reading 'cur' before it is re-filled
    }

    // store: C layout m = r + 8*(lane>>4), n = lane&15 ; only N is ragged
    #pragma unroll
    for (int i = 0; i < 4; ++i)
        #pragma unroll
        for (int j = 0; j < 2; ++j) {
            int mb = m0 + wm + i * 16 + 8 * lh;
            int nb = n0 + wn + j * 16 + ll;
            if (nb < N) {
                #pragma unroll
                for (int r = 0; r < 8; ++r) {
                    float v = acc[i][j][r];
                    if (mode == 1) {
                        v += bias[nb];
                        v = (v > 20.f) ? v : log1pf(__expf(v));
                    }
                    size_t off = (size_t)(mb + r) * ldc + nb;
                    C[off] = v;
                    if (C2) C2[off] = f2bf(v);
                }
            }
        }
}

// ---------------------------------------------------------------------------
// Depthwise causal conv1d (k=4) + bias + SiLU; writes fp32 and bf16 copies
// ---------------------------------------------------------------------------
__global__ __launch_bounds__(256)
void conv_silu_kernel(const float* __restrict__ xz,
                      const float* __restrict__ cw,
                      const float* __restrict__ cb,
                      float* __restrict__ xl, bf16* __restrict__ xlb,
                      int Bt, int L)
{
    const int DI = 2048;
    int i = blockIdx.x * blockDim.x + threadIdx.x;
    if (i >= Bt * L * DI) return;
    int d = i % DI;
    int t = (i / DI) % L;
    int b = i / (DI * L);
    float acc = cb[d];
    #pragma unroll
    for (int j = 0; j < 4; ++j) {
        int ts = t - 3 + j;
        if (ts >= 0)
            acc += xz[((size_t)(b * L + ts)) * 4096 + d] * cw[d * 4 + j];
    }
    float s = silu_f(acc);
    size_t o = ((size_t)(b * L + t)) * DI + d;
    xl[o]  = s;
    xlb[o] = f2bf(s);
}

// ---------------------------------------------------------------------------
// Selective scan: thread per (b, channel); 16 states in registers; B/C vectors
// LDS-staged per 64-step tile; fused D-skip + SiLU(z) gate; bf16 output.
// ---------------------------------------------------------------------------
__global__ __launch_bounds__(256)
void scan_kernel(const float* __restrict__ xl,
                 const float* __restrict__ delta,
                 const float* __restrict__ xdbl,     // [M,96]; B@64, C@80
                 const float* __restrict__ xz,       // z = xz[:, 2048:]
                 const float* __restrict__ A_log,
                 const float* __restrict__ Dp,
                 bf16* __restrict__ ybf, int L)
{
    const int DI = 2048, NS = 16, TT = 64;
    __shared__ float sB[TT][NS];
    __shared__ float sC[TT][NS];

    int tid = threadIdx.x;
    int b   = blockIdx.x / (DI / 256);
    int d   = (blockIdx.x % (DI / 256)) * 256 + tid;

    float Arow[NS], h[NS];
    #pragma unroll
    for (int n = 0; n < NS; ++n) {
        Arow[n] = -__expf(A_log[d * NS + n]);
        h[n] = 0.f;
    }
    const float Dd = Dp[d];

    for (int t0 = 0; t0 < L; t0 += TT) {
        __syncthreads();
        #pragma unroll
        for (int s = 0; s < (TT * NS) / 256; ++s) {
            int idx = tid + s * 256;
            int r = idx >> 4, c = idx & 15;
            size_t row = (size_t)(b * L + t0 + r) * 96;
            sB[r][c] = xdbl[row + 64 + c];
            sC[r][c] = xdbl[row + 80 + c];
        }
        __syncthreads();

        for (int tt = 0; tt < TT; ++tt) {
            size_t base = (size_t)(b * L + t0 + tt);
            float u  = xl[base * DI + d];
            float dl = delta[base * DI + d];
            float du = dl * u;
            float y = 0.f;
            #pragma unroll
            for (int n = 0; n < NS; ++n) {
                float dA = __expf(dl * Arow[n]);
                h[n] = dA * h[n] + du * sB[tt][n];
                y += h[n] * sC[tt][n];
            }
            float zv = xz[base * 4096 + 2048 + d];
            ybf[base * DI + d] = f2bf((y + u * Dd) * silu_f(zv));
        }
    }
}

// ---------------------------------------------------------------------------
extern "C" void kernel_launch(void* const* d_in, const int* in_sizes, int n_in,
                              void* d_out, int out_size, void* d_ws, size_t ws_size,
                              hipStream_t stream) {
    const float* hs   = (const float*)d_in[0];  // [2,2048,1024]
    const float* inw  = (const float*)d_in[1];  // [4096,1024]
    const float* cw   = (const float*)d_in[2];  // [2048,1,4]
    const float* cb   = (const float*)d_in[3];  // [2048]
    const float* xpw  = (const float*)d_in[4];  // [96,2048]
    const float* dtw  = (const float*)d_in[5];  // [2048,64]
    const float* dtb  = (const float*)d_in[6];  // [2048]
    const float* alog = (const float*)d_in[7];  // [2048,16]
    const float* Dpp  = (const float*)d_in[8];  // [2048]
    const float* opw  = (const float*)d_in[9];  // [1024,2048]
    float* out = (float*)d_out;

    const int Bt = 2, L = 2048, DM = 1024, DI = 2048;
    const int M = Bt * L;                        // 4096

    // ---- workspace layout (bytes) ----
    char* ws = (char*)d_ws;
    size_t off = 0;
    float* xz      = (float*)(ws + off); off += (size_t)M * 4096 * 4;     // 64MB
    float* xl      = (float*)(ws + off); off += (size_t)M * DI * 4;       // 32MB
    float* delta   = (float*)(ws + off); off += (size_t)M * DI * 4;       // 32MB
    float* xdbl    = (float*)(ws + off); off += (size_t)M * 96 * 4;
    bf16*  hs_bf   = (bf16*)(ws + off);  off += (size_t)M * DM * 2;
    bf16*  inw_bf  = (bf16*)(ws + off);  off += (size_t)4096 * DM * 2;
    bf16*  opw_bf  = (bf16*)(ws + off);  off += (size_t)DM * DI * 2;
    bf16*  xpw_bf  = (bf16*)(ws + off);  off += (size_t)128 * DI * 2;     // padded to 128 rows
    bf16*  dtw_bf  = (bf16*)(ws + off);  off += (size_t)DI * 64 * 2;
    bf16*  xl_bf   = (bf16*)(ws + off);  off += (size_t)M * DI * 2;
    bf16*  xdbl_bf = (bf16*)(ws + off);  off += (size_t)M * 96 * 2;
    bf16*  y_bf    = (bf16*)(ws + off);  off += (size_t)M * DI * 2;

    // ---- 0) weight/activation conversions to bf16 ----
    auto cvt = [&](const float* s, bf16* d, size_t n) {
        int n4 = (int)(n / 4);
        cvt_f32_to_bf16<<<(n4 + 255) / 256, 256, 0, stream>>>(s, (unsigned*)d, n4);
    };
    cvt(hs,  hs_bf,  (size_t)M * DM);
    cvt(inw, inw_bf, (size_t)4096 * DM);
    cvt(xpw, xpw_bf, (size_t)96 * DI);
    cvt(dtw, dtw_bf, (size_t)DI * 64);
    cvt(opw, opw_bf, (size_t)DM * DI);

    dim3 blk(256);

    // 1) xz = hs @ in_proj_w^T                     [M x 4096]
    gemm_bf16_wmma<<<dim3(4096 / BN, M / BM), blk, 0, stream>>>(
        hs_bf, DM, inw_bf, DM, xz, 4096, nullptr, M, 2 * DI, DM, nullptr, 0);

    // 2) xl = SiLU(conv(x) + b)   (fp32 + bf16)
    int tot = M * DI;
    conv_silu_kernel<<<(tot + 255) / 256, 256, 0, stream>>>(
        xz, cw, cb, xl, xl_bf, Bt, L);

    // 3) x_dbl = xl @ x_proj_w^T                   [M x 96] (fp32 + bf16)
    gemm_bf16_wmma<<<dim3((96 + BN - 1) / BN, M / BM), blk, 0, stream>>>(
        xl_bf, DI, xpw_bf, DI, xdbl, 96, xdbl_bf, M, 96, DI, nullptr, 0);

    // 4) delta = softplus(x_dbl[:, :64] @ dt_proj_w^T + b)   [M x 2048]
    gemm_bf16_wmma<<<dim3(DI / BN, M / BM), blk, 0, stream>>>(
        xdbl_bf, 96, dtw_bf, 64, delta, DI, nullptr, M, DI, 64, dtb, 1);

    // 5) selective scan + D-skip + SiLU(z) gate -> y_bf
    scan_kernel<<<Bt * (DI / 256), 256, 0, stream>>>(
        xl, delta, xdbl, xz, alog, Dpp, y_bf, L);

    // 6) out = y @ out_proj_w^T                    [M x 1024]
    gemm_bf16_wmma<<<dim3(DM / BN, M / BM), blk, 0, stream>>>(
        y_bf, DI, opw_bf, DI, out, DM, nullptr, M, DM, DI, nullptr, 0);
}
